// NMS_12335146074352
// MI455X (gfx1250) — 8-bit matrix operand order repaired and checked by reading, compiled, not run on hardware
//
#include <hip/hip_runtime.h>
#include <hip/hip_bf16.h>
#include <float.h>

#ifndef __has_builtin
#define __has_builtin(x) 0
#endif

#define BATCH 32
#define NQ    10000
#define NC    80
#define TOPK  200
#define QPAD  84
#define K1_THREADS 128
#define NPAD  208
#define NTILE 13
#define TK_THREADS 512
#define NBIN  2048
#define CCAP  2048

typedef __attribute__((ext_vector_type(8))) int v8i;
typedef int v4i_t __attribute__((vector_size(16)));

#define AS1 __attribute__((address_space(1)))
#define AS3 __attribute__((address_space(3)))

__device__ __forceinline__ void async_b128_to_lds(const void* g, void* l) {
#if __has_builtin(__builtin_amdgcn_global_load_async_to_lds_b128)
  __builtin_amdgcn_global_load_async_to_lds_b128((AS1 v4i_t*)g, (AS3 v4i_t*)l, 0, 0);
#else
  unsigned lds_addr = (unsigned)(unsigned long long)(AS3 char*)l;
  asm volatile("global_load_async_to_lds_b128 %0, %1, off"
               :: "v"(lds_addr), "v"((unsigned long long)g) : "memory");
#endif
}

__device__ __forceinline__ void wait_asynccnt0() {
#if __has_builtin(__builtin_amdgcn_s_wait_asynccnt)
  __builtin_amdgcn_s_wait_asynccnt(0);
#else
  asm volatile("s_wait_asynccnt 0" ::: "memory");
#endif
}

__global__ __launch_bounds__(K1_THREADS)
void score_kernel(const float* __restrict__ logits,
                  float* __restrict__ scores,
                  int* __restrict__ classes) {
  __shared__ float sbuf[K1_THREADS * QPAD];
  const int tid = threadIdx.x;
  const int q = blockIdx.x * K1_THREADS + tid;
  const float* g = logits + (size_t)q * NC;
  float* l = &sbuf[tid * QPAD];
#pragma unroll
  for (int j = 0; j < NC / 4; ++j)
    async_b128_to_lds(g + j * 4, l + j * 4);
  wait_asynccnt0();

  float m = -FLT_MAX; int am = 0;
#pragma unroll 4
  for (int i = 0; i < NC; ++i) {
    float v = l[i];
    if (v > m) { m = v; am = i; }
  }
  float s = 0.f;
#pragma unroll 4
  for (int i = 0; i < NC; ++i)
    s += __expf(l[i] - m);
  scores[q]  = 1.0f / s;
  classes[q] = am;
}

__global__ __launch_bounds__(TK_THREADS)
void topk_kernel(const float* __restrict__ scores, int* __restrict__ order) {
  __shared__ unsigned s_hist[NBIN];
  __shared__ unsigned long long s_cand[CCAP];
  __shared__ unsigned s_cnt;
  __shared__ unsigned s_thr;
  const int b = blockIdx.x, tid = threadIdx.x;
  const float* sc = scores + b * NQ;

  for (int i = tid; i < NBIN; i += TK_THREADS) s_hist[i] = 0u;
  if (tid == 0) s_cnt = 0u;
  __syncthreads();

  for (int i = tid; i < NQ; i += TK_THREADS) {
    unsigned bits = __float_as_uint(sc[i]);
    atomicAdd(&s_hist[bits >> 20], 1u);
  }
  __syncthreads();

  if (tid == 0) {
    unsigned cum = 0; int bstar = 0;
    for (int bin = NBIN - 1; bin >= 0; --bin) {
      cum += s_hist[bin];
      if (cum >= TOPK) { bstar = bin; break; }
    }
    s_thr = (unsigned)bstar;
  }
  __syncthreads();
  const unsigned thr = s_thr;

  for (int i = tid; i < NQ; i += TK_THREADS) {
    unsigned bits = __float_as_uint(sc[i]);
    if ((bits >> 20) >= thr) {
      unsigned slot = atomicAdd(&s_cnt, 1u);
      if (slot < CCAP)
        s_cand[slot] = ((unsigned long long)bits << 32) |
                       (unsigned long long)(unsigned)(NQ - 1 - i);
    }
  }
  __syncthreads();
  const int C = (s_cnt < (unsigned)CCAP) ? (int)s_cnt : CCAP;

  if (tid < 32) {
    const int lane = tid;
    for (int k = 0; k < TOPK; ++k) {
      unsigned long long best = 0ull; int bslot = -1;
      for (int i = lane; i < C; i += 32) {
        unsigned long long v = s_cand[i];
        if (v > best) { best = v; bslot = i; }
      }
#pragma unroll
      for (int off = 16; off > 0; off >>= 1) {
        unsigned long long ov = __shfl_xor(best, off, 32);
        int os = __shfl_xor(bslot, off, 32);
        if (ov > best) { best = ov; bslot = os; }
      }
      if (lane == 0) {
        order[b * TOPK + k] = NQ - 1 - (int)(unsigned)(best & 0xffffffffu);
        if (bslot >= 0) s_cand[bslot] = 0ull;
      }
    }
  }
}

__global__ __launch_bounds__(256)
void nms_kernel(const int* __restrict__ order,
                const int* __restrict__ classes,
                const float* __restrict__ seg,
                int* __restrict__ out) {
  __shared__ int      s_cls[NPAD];
  __shared__ float    s_x1[NPAD], s_x2[NPAD];
  __shared__ int      s_ord[NPAD];
  __shared__ unsigned s_sup[TOPK * 8];
  __shared__ int      s_keep[TOPK];
  const int b = blockIdx.x, t = threadIdx.x;

  if (t < NPAD) {
    if (t < TOPK) {
      int o = order[b * TOPK + t];
      s_ord[t] = o;
      s_cls[t] = classes[b * NQ + o];
      s_x1[t]  = seg[((size_t)b * NQ + o) * 2 + 0];
      s_x2[t]  = seg[((size_t)b * NQ + o) * 2 + 1];
    } else {
      s_ord[t] = -1; s_cls[t] = -1;
      s_x1[t] = 0.f; s_x2[t] = 0.f;
    }
  }
  for (int i = t; i < TOPK * 8; i += 256) s_sup[i] = 0u;
  if (t < TOPK) s_keep[t] = -1;
  __syncthreads();

  const int wave = t >> 5, lane = t & 31;
  const int hl = lane >> 4, l16 = lane & 15;
  for (int ti = wave; ti < NTILE; ti += 8) {
    const int ci = s_cls[ti * 16 + l16];
    v8i a0, a1;
#pragma unroll
    for (int v = 0; v < 8; ++v) {
      const int kbA = 32 * (v >> 2) + 16 * ((v >> 1) & 1) + 8 * hl + 4 * (v & 1);
      unsigned wa0 = 0, wa1 = 0;
#pragma unroll
      for (int bb = 0; bb < 4; ++bb) {
        wa0 |= (ci == kbA + bb)      ? (1u << (8 * bb)) : 0u;
        wa1 |= (ci == kbA + bb + 64) ? (1u << (8 * bb)) : 0u;
      }
      a0[v] = (int)wa0; a1[v] = (int)wa1;
    }
    for (int tj = 0; tj < NTILE; ++tj) {
      const int cj = s_cls[tj * 16 + l16];
      v8i b0, b1;
#pragma unroll
      for (int v = 0; v < 8; ++v) {
        const int kbB = 32 * (v >> 2) + 16 * ((v >> 1) & 1) + 8 * hl + 4 * (v & 1);
        unsigned wb0 = 0, wb1 = 0;
#pragma unroll
        for (int bb = 0; bb < 4; ++bb) {
          wb0 |= (cj == kbB + bb)      ? (1u << (8 * bb)) : 0u;
          wb1 |= (cj == kbB + bb + 64) ? (1u << (8 * bb)) : 0u;
        }
        b0[v] = (int)wb0; b1[v] = (int)wb1;
      }
      v8i acc = {0, 0, 0, 0, 0, 0, 0, 0};
      acc = __builtin_amdgcn_wmma_i32_16x16x64_iu8(false, a0, false, b0, acc, false, false);
      acc = __builtin_amdgcn_wmma_i32_16x16x64_iu8(false, a1, false, b1, acc, false, false);
#pragma unroll
      for (int r = 0; r < 8; ++r) {
        const int i = ti * 16 + r + 8 * hl;
        const int j = tj * 16 + l16;
        if (i < TOPK && j < TOPK && j > i && acc[r] != 0) {
          const float inter = fminf(s_x2[i], s_x2[j]) - fmaxf(s_x1[i], s_x1[j]);
          if (inter > 0.0f)
            atomicOr(&s_sup[i * 8 + (j >> 5)], 1u << (j & 31));
        }
      }
    }
  }
  __syncthreads();

  if (t == 0) {
    unsigned act[7];
#pragma unroll
    for (int w = 0; w < 7; ++w) act[w] = 0xffffffffu;
    int cnt = 0;
    for (int p = 0; p < TOPK; ++p) {
      if ((act[p >> 5] >> (p & 31)) & 1u) {
        s_keep[cnt++] = s_ord[p];
#pragma unroll
        for (int w = 0; w < 7; ++w) act[w] &= ~s_sup[p * 8 + w];
      }
    }
  }
  __syncthreads();
  if (t < TOPK) {
    out[b * TOPK + t] = s_keep[t];
    out[(BATCH + b) * TOPK + t] = 0;
  }
}

extern "C" void kernel_launch(void* const* d_in, const int* in_sizes, int n_in,
                              void* d_out, int out_size, void* d_ws, size_t ws_size,
                              hipStream_t stream) {
  (void)in_sizes; (void)n_in; (void)out_size; (void)ws_size;
  const float* logits = (const float*)d_in[0];
  const float* seg    = (const float*)d_in[1];
  int* out = (int*)d_out;

  char* ws = (char*)d_ws;
  float* scores = (float*)ws;
  int*   classes = (int*)(ws + (size_t)BATCH * NQ * 4);
  int*   order   = (int*)(ws + (size_t)BATCH * NQ * 8);

  score_kernel<<<(BATCH * NQ) / K1_THREADS, K1_THREADS, 0, stream>>>(logits, scores, classes);
  topk_kernel<<<BATCH, TK_THREADS, 0, stream>>>(scores, order);
  nms_kernel<<<BATCH, 256, 0, stream>>>(order, classes, seg, out);
}
